// SimpleGRU_17566416240914
// MI455X (gfx1250) — compile-verified, matrix operand order checked by reading, not run on hardware
//
#include <hip/hip_runtime.h>
#include <cstdint>

typedef _Float16 half_t;
typedef __attribute__((ext_vector_type(16))) _Float16 v16h;
typedef __attribute__((ext_vector_type(8)))  float    v8f;

#define B_  256
#define T_  256
#define D_  128
#define H_  512
#define G_  1536   // 3*H
#define NS_ 24

// ---------------- WMMA fragment helpers (wave32, 16x16x32 f16) ----------------
// A 16x32 f16 layout (ISA 7.12.2): lane m=l&15; lanes>=16 shift K by 8;
// VGPR v holds K = (v>=4?16:0) + kb + 2*(v&3) + {0,1}.
__device__ inline v16h load_a_frag(const half_t* __restrict__ base, int row0, int ld,
                                   int kk, int lane) {
  const int m  = lane & 15;
  const int kb = (lane >> 4) << 3;  // 0 or 8
  const half_t* p = base + (size_t)(row0 + m) * ld + kk + kb;
  v16h a;
#pragma unroll
  for (int v = 0; v < 8; ++v) {
    const int k0 = ((v >> 2) << 4) + ((v & 3) << 1);
    union { uint32_t u; half_t h[2]; } cv;
    cv.u = *reinterpret_cast<const uint32_t*>(p + k0);   // vectorizes to b128 pairs
    a[2 * v]     = cv.h[0];
    a[2 * v + 1] = cv.h[1];
  }
  return a;
}

// B fragments pre-packed (pack_w_kernel) into per-lane contiguous 32B blocks:
// halves address = ((kt*96 + ntile)*32 + lane)*16  ->  two global_load_b128 per lane.
__device__ inline v16h load_b_frag(const half_t* __restrict__ wpk, int kt, int ntile, int lane) {
  return *reinterpret_cast<const v16h*>(
      wpk + (((size_t)kt * (G_ / 16) + ntile) * 32 + lane) * 16);
}

__device__ inline v8f wmma16(v16h a, v16h b, v8f c) {
  return __builtin_amdgcn_wmma_f32_16x16x32_f16(false, a, false, b, (short)0, c, false, false);
}

// ---------------- weight repack: fp32 [G rows, K cols] -> f16 fragment order ----------------
__global__ void pack_w_kernel(half_t* __restrict__ dst, const float* __restrict__ src,
                              int K, int total) {
  int p = blockIdx.x * blockDim.x + threadIdx.x;
  if (p >= total) return;
  int e    = p & 15;
  int lane = (p >> 4) & 31;
  int rest = p >> 9;
  int ntile = rest % (G_ / 16);
  int kt    = rest / (G_ / 16);
  int v = e >> 1, j = e & 1;
  int n = ntile * 16 + (lane & 15);
  int k = kt * 32 + ((v >> 2) << 4) + (((lane >> 4) & 1) << 3) + ((v & 3) << 1) + j;
  dst[p] = (half_t)src[(size_t)n * K + k];   // B[k][n] = w[n][k] (w^T)
}

__global__ void cvt_f32_to_f16_kernel(half_t* __restrict__ dst, const float* __restrict__ src, int n) {
  int i = blockIdx.x * blockDim.x + threadIdx.x;
  if (i < n) dst[i] = (half_t)src[i];
}

__global__ void zero_h_kernel(half_t* __restrict__ h16, float* __restrict__ h32, int n) {
  int i = blockIdx.x * blockDim.x + threadIdx.x;
  if (i < n) { h16[i] = (half_t)0.f; h32[i] = 0.f; }
}

// ---------------- input-gate GEMM: gx[bt,g] = x[bt,:] @ w_ih^T + b_ih ----------------
// One wave owns a 16x64 output strip (A-fragment reuse over 4 N-tiles).
__global__ void __launch_bounds__(256) gemm_in_kernel(
    half_t* __restrict__ gx, const half_t* __restrict__ A, const half_t* __restrict__ wpk,
    const float* __restrict__ b_ih, int K) {
  const int wave   = (blockIdx.x * blockDim.x + threadIdx.x) >> 5;
  const int lane   = threadIdx.x & 31;
  const int nstrip = wave % (G_ / 64);   // 24 strips of 64 cols
  const int mtile  = wave / (G_ / 64);   // 4096 row tiles
  const int row0   = mtile * 16;
  const int nt0    = nstrip * 4;

  v8f c0 = {}, c1 = {}, c2 = {}, c3 = {};
  const int kts = K >> 5;
  for (int kt = 0; kt < kts; ++kt) {
    v16h a  = load_a_frag(A, row0, K, kt << 5, lane);   // ld == K for every layer input
    v16h b0 = load_b_frag(wpk, kt, nt0 + 0, lane);
    v16h b1 = load_b_frag(wpk, kt, nt0 + 1, lane);
    v16h b2 = load_b_frag(wpk, kt, nt0 + 2, lane);
    v16h b3 = load_b_frag(wpk, kt, nt0 + 3, lane);
    c0 = wmma16(a, b0, c0);
    c1 = wmma16(a, b1, c1);
    c2 = wmma16(a, b2, c2);
    c3 = wmma16(a, b3, c3);
  }

  const int n  = lane & 15;
  const int mh = (lane >> 4) << 3;
#pragma unroll
  for (int jj = 0; jj < 4; ++jj) {
    const v8f& c = (jj == 0) ? c0 : (jj == 1) ? c1 : (jj == 2) ? c2 : c3;
    const int col   = (nt0 + jj) * 16 + n;
    const float bia = b_ih[col];
#pragma unroll
    for (int r = 0; r < 8; ++r) {
      const int bt = row0 + mh + r;
      gx[(size_t)bt * G_ + col] = (half_t)(c[r] + bia);
    }
  }
}

// ---------------- one recurrent time-step, all gates fused ----------------
// Block = 4 waves sharing one htile (identical B fragments): stage all 48KB of
// this htile's w_hh fragments into LDS once via async global->LDS (ASYNCcnt),
// then all 4 waves feed WMMA from ds_load_b128. 128 blocks x 128 threads.
__global__ void __launch_bounds__(128) gru_step_kernel(
    const half_t* __restrict__ h16_in, const float* __restrict__ h32_in,
    half_t* __restrict__ h16_out, float* __restrict__ h32_out,
    half_t* __restrict__ y16, const half_t* __restrict__ gx,
    const half_t* __restrict__ wpk_hh, const float* __restrict__ b_hh, int t) {
  const int lane  = threadIdx.x & 31;
  const int wavei = threadIdx.x >> 5;            // 0..3
  const int htile = blockIdx.x & 31;
  const int btile = ((blockIdx.x >> 5) << 2) + wavei;  // 0..15
  const int row0  = btile * 16;

  // [g][kt][lane][16] halves = 3*16*32*16*2 = 48KB
  __shared__ __align__(32) half_t wlds[3 * 16 * 32 * 16];

  // ---- async global->LDS stage: 3072 x b128 transfers, 24 per thread ----
  {
    const int tid = threadIdx.x;
    const uint32_t lds0 = (uint32_t)(uintptr_t)(&wlds[0]);   // low 32 bits = LDS byte addr
    const uint64_t wsrc = (uint64_t)(uintptr_t)wpk_hh;
#pragma unroll
    for (int j = 0; j < 24; ++j) {
      const int id  = j * 128 + tid;
      const int blk = id >> 6;            // 0..47 = g*16 + kt
      const int wb  = (id & 63) << 4;     // byte offset within 1KB fragment block
      const int g   = blk >> 4;
      const int kt  = blk & 15;
      const uint64_t src = wsrc + (uint64_t)(((kt * 96 + g * 32 + htile) << 10) + wb);
      const uint32_t dst = lds0 + (uint32_t)((blk << 10) + wb);
      asm volatile("global_load_async_to_lds_b128 %0, %1, off"
                   :: "v"(dst), "v"(src) : "memory");
    }
    asm volatile("s_wait_asynccnt 0" ::: "memory");
    __syncthreads();
  }

  v8f cr = {}, cz = {}, cn = {};
#pragma unroll
  for (int kt = 0; kt < 16; ++kt) {
    v16h a  = load_a_frag(h16_in, row0, H_, kt << 5, lane);
    v16h br = *reinterpret_cast<const v16h*>(&wlds[((0 * 16 + kt) * 32 + lane) << 4]);
    v16h bz = *reinterpret_cast<const v16h*>(&wlds[((1 * 16 + kt) * 32 + lane) << 4]);
    v16h bn = *reinterpret_cast<const v16h*>(&wlds[((2 * 16 + kt) * 32 + lane) << 4]);
    cr = wmma16(a, br, cr);
    cz = wmma16(a, bz, cz);
    cn = wmma16(a, bn, cn);
  }

  const int n   = lane & 15;
  const int mh  = (lane >> 4) << 3;
  const int col = htile * 16 + n;
  const float bhr = b_hh[col];
  const float bhz = b_hh[H_ + col];
  const float bhn = b_hh[2 * H_ + col];

#pragma unroll
  for (int r = 0; r < 8; ++r) {
    const int b = row0 + mh + r;
    const size_t grow = (size_t)(b * T_ + t) * G_;
    const float xr = (float)gx[grow + col];
    const float xz = (float)gx[grow + H_ + col];
    const float xn = (float)gx[grow + 2 * H_ + col];
    const float rg = 1.f / (1.f + expf(-(xr + cr[r] + bhr)));
    const float zg = 1.f / (1.f + expf(-(xz + cz[r] + bhz)));
    const float ng = tanhf(xn + rg * (cn[r] + bhn));
    const float hp = h32_in[(size_t)b * H_ + col];
    const float hv = (1.f - zg) * ng + zg * hp;
    h32_out[(size_t)b * H_ + col] = hv;                 // fp32 carry state
    h16_out[(size_t)b * H_ + col] = (half_t)hv;         // f16 copy for next A-fragments
    y16[(size_t)(b * T_ + t) * H_ + col] = (half_t)hv;  // layer output
  }
}

// ---------------- tiny final FC (256x24, fp32) ----------------
__global__ void fc_kernel(float* __restrict__ out, const float* __restrict__ h,
                          const float* __restrict__ w, const float* __restrict__ bv) {
  int i = blockIdx.x * blockDim.x + threadIdx.x;
  if (i >= B_ * NS_) return;
  int b = i / NS_, s = i % NS_;
  float acc = bv[s];
  for (int k = 0; k < H_; ++k) acc += h[(size_t)b * H_ + k] * w[(size_t)s * H_ + k];
  out[i] = acc;
}

extern "C" void kernel_launch(void* const* d_in, const int* in_sizes, int n_in,
                              void* d_out, int out_size, void* d_ws, size_t ws_size,
                              hipStream_t stream) {
  (void)in_sizes; (void)n_in; (void)out_size; (void)ws_size;
  const float* x = (const float*)d_in[0];
  const float* w_ih[3] = {(const float*)d_in[1], (const float*)d_in[5], (const float*)d_in[9]};
  const float* w_hh[3] = {(const float*)d_in[2], (const float*)d_in[6], (const float*)d_in[10]};
  const float* b_ih[3] = {(const float*)d_in[3], (const float*)d_in[7], (const float*)d_in[11]};
  const float* b_hh[3] = {(const float*)d_in[4], (const float*)d_in[8], (const float*)d_in[12]};
  const float* fc_w = (const float*)d_in[13];
  const float* fc_b = (const float*)d_in[14];

  char* ws = (char*)d_ws;
  size_t off = 0;
  auto alloc = [&](size_t bytes) -> void* {
    void* p = ws + off;
    off = (off + bytes + 255) & ~(size_t)255;
    return p;
  };
  half_t* xin16 = (half_t*)alloc((size_t)B_ * T_ * D_ * 2);
  half_t* ybufA = (half_t*)alloc((size_t)B_ * T_ * H_ * 2);
  half_t* ybufB = (half_t*)alloc((size_t)B_ * T_ * H_ * 2);
  half_t* gx16  = (half_t*)alloc((size_t)B_ * T_ * G_ * 2);
  half_t* wpk_ih[3]; half_t* wpk_hh[3];
  wpk_ih[0] = (half_t*)alloc((size_t)G_ * D_ * 2);
  wpk_ih[1] = (half_t*)alloc((size_t)G_ * H_ * 2);
  wpk_ih[2] = (half_t*)alloc((size_t)G_ * H_ * 2);
  for (int l = 0; l < 3; ++l) wpk_hh[l] = (half_t*)alloc((size_t)G_ * H_ * 2);
  half_t* h16b[2]; float* h32b[2];
  h16b[0] = (half_t*)alloc((size_t)B_ * H_ * 2);
  h16b[1] = (half_t*)alloc((size_t)B_ * H_ * 2);
  h32b[0] = (float*)alloc((size_t)B_ * H_ * 4);
  h32b[1] = (float*)alloc((size_t)B_ * H_ * 4);

  // Repack weights into WMMA fragment order (once per call).
  const int Kih[3] = {D_, H_, H_};
  for (int l = 0; l < 3; ++l) {
    int tot = G_ * Kih[l];
    pack_w_kernel<<<(tot + 255) / 256, 256, 0, stream>>>(wpk_ih[l], w_ih[l], Kih[l], tot);
    int tot2 = G_ * H_;
    pack_w_kernel<<<(tot2 + 255) / 256, 256, 0, stream>>>(wpk_hh[l], w_hh[l], H_, tot2);
  }
  {
    int n = B_ * T_ * D_;
    cvt_f32_to_f16_kernel<<<(n + 255) / 256, 256, 0, stream>>>(xin16, x, n);
  }

  const half_t* layer_in[3]  = {xin16, ybufA, ybufB};
  half_t*       layer_out[3] = {ybufA, ybufB, ybufA};
  const int     Kl[3]        = {D_, H_, H_};

  for (int l = 0; l < 3; ++l) {
    // 1) precompute input gates for the whole layer: 98304 waves, 16x64 tiles each
    gemm_in_kernel<<<12288, 256, 0, stream>>>(gx16, layer_in[l], wpk_ih[l], b_ih[l], Kl[l]);
    // 2) h0 = 0
    zero_h_kernel<<<(B_ * H_ + 255) / 256, 256, 0, stream>>>(h16b[0], h32b[0], B_ * H_);
    // 3) serial scan; ping-pong h buffers, kernel boundary = step barrier
    for (int t = 0; t < T_; ++t) {
      int pi = t & 1, po = (t + 1) & 1;
      gru_step_kernel<<<128, 128, 0, stream>>>(h16b[pi], h32b[pi], h16b[po], h32b[po],
                                               layer_out[l], gx16, wpk_hh[l], b_hh[l], t);
    }
  }
  // T_ even -> final state landed in bank 0 (fp32).
  fc_kernel<<<(B_ * NS_ + 255) / 256, 256, 0, stream>>>((float*)d_out, h32b[0], fc_w, fc_b);
}
